// MicroDiseaseModel_29188597743707
// MI455X (gfx1250) — matrix operand order repaired and checked by reading, not compiled, not run on hardware
//
#include <hip/hip_runtime.h>
#include <hip/hip_bf16.h>
#include <hip/hip_fp16.h>

typedef __attribute__((ext_vector_type(16))) _Float16 v16h;
typedef __attribute__((ext_vector_type(8)))  _Float16 v8h;
typedef __attribute__((ext_vector_type(8)))  float    v8f;

#define BM 128
#define BN 64
#define BK 32
#define LDT 40   // padded LDS row stride in halves (80 B, 16B-aligned rows)

// ---------------------------------------------------------------------------
// CDNA5 async global->LDS DMA (ASYNCcnt-tracked). Probe round 2 showed the
// builtin exists with param0 = 'int __vector(4) __device__ *' (AS1, non-const).
// ---------------------------------------------------------------------------
#if defined(__HIP_DEVICE_COMPILE__) && __has_builtin(__builtin_amdgcn_global_load_async_to_lds_b128)
#define HAVE_ASYNC_LDS 1
#else
#define HAVE_ASYNC_LDS 0
#endif

#if HAVE_ASYNC_LDS
typedef int v4i_vs __attribute__((vector_size(16)));
typedef __attribute__((address_space(1))) v4i_vs* as1_v4i_p;
typedef __attribute__((address_space(3))) v4i_vs* as3_v4i_p;

__device__ __forceinline__ void async_copy_b128(const void* gsrc, void* ldst)
{
    __builtin_amdgcn_global_load_async_to_lds_b128(
        (as1_v4i_p)(uintptr_t)gsrc,
        (as3_v4i_p)(uint32_t)(uintptr_t)ldst,
        0, 0);
}
#endif

template<int N>
__device__ __forceinline__ void wait_asynccnt()
{
#if defined(__HIP_DEVICE_COMPILE__)
#if __has_builtin(__builtin_amdgcn_s_wait_asynccnt)
    __builtin_amdgcn_s_wait_asynccnt(N);
#else
    asm volatile("s_wait_asynccnt %0" :: "i"(N) : "memory");
#endif
#endif
}

// ---------------------------------------------------------------------------
// f32 -> f16 conversion, 8 elements / thread
// ---------------------------------------------------------------------------
__global__ __launch_bounds__(256) void f32_to_f16_kernel(
    const float* __restrict__ in, _Float16* __restrict__ out, int n8)
{
    int i = blockIdx.x * 256 + threadIdx.x;
    if (i >= n8) return;
    const float4 a = ((const float4*)in)[i * 2 + 0];
    const float4 b = ((const float4*)in)[i * 2 + 1];
    v8h h;
    h[0] = (_Float16)a.x; h[1] = (_Float16)a.y;
    h[2] = (_Float16)a.z; h[3] = (_Float16)a.w;
    h[4] = (_Float16)b.x; h[5] = (_Float16)b.y;
    h[6] = (_Float16)b.z; h[7] = (_Float16)b.w;
    *(v8h*)&out[(size_t)i * 8] = h;
}

// ---------------------------------------------------------------------------
// NT GEMM: C[M,N] = act(A[M,K] @ W[N,K]^T + bias[N]),  K/N compile-time.
// 8 waves/block, 128x64 block tile, 32x32 wave tile (2x2 WMMA f16->f32),
// K-step 32. Async path: 4-stage LDS pipeline, tiles issued 2 steps ahead,
// one barrier per K-step. Fallback: sync ping-pong (2 stages, 1 barrier).
// ---------------------------------------------------------------------------
template<int KD, int ND, bool RELU, typename OutT>
__global__ __launch_bounds__(256) void gemm_bias_act(
    const _Float16* __restrict__ A,   // M x KD row-major
    const _Float16* __restrict__ Wt,  // ND x KD row-major (B = Wt^T, K-major)
    const float*    __restrict__ bias,// ND
    OutT*           __restrict__ C,   // M x ND row-major
    int M)
{
    static_assert(KD % BK == 0 && KD >= 2 * BK, "K tiling");
#if HAVE_ASYNC_LDS
    constexpr int NSTAGE = 4;
#else
    constexpr int NSTAGE = 2;
#endif
    __shared__ _Float16 lA[NSTAGE][BM * LDT];
    __shared__ _Float16 lB[NSTAGE][BN * LDT];

    const int tid   = threadIdx.x;
    const int lane  = tid & 31;
    const int wave  = tid >> 5;
    const int wm    = wave & 3;     // 4 x 32 = 128 M
    const int wn    = wave >> 2;    // 2 x 32 =  64 N
    const int m0    = blockIdx.y * BM;
    const int n0    = blockIdx.x * BN;
    const int l16   = lane & 15;
    const int lhalf = lane >> 4;

    // one fixed 16B staging chunk set per thread; advance by BK halves per step
    const int ar0 = tid >> 2;                 // A rows 0..63
    const int ar1 = ar0 + 64;                 // A rows 64..127
    const int ac  = (tid & 3) * 8;            // chunk column (halves)
    const _Float16* gA0 = A  + (size_t)(m0 + ar0) * KD + ac;
    const _Float16* gA1 = A  + (size_t)(m0 + ar1) * KD + ac;
    const _Float16* gB  = Wt + (size_t)(n0 + ar0) * KD + ac;   // B rows 0..63

    v8f acc[2][2] = {};

    auto compute_tile = [&](const _Float16* sA, const _Float16* sB) {
        v16h af[2], bf[2];
#pragma unroll
        for (int t = 0; t < 2; ++t) {
            // A 16x32 f16 layout: lane holds K = {8*lhalf..+7, 16+8*lhalf..+7}
            int arow = wm * 32 + t * 16 + l16;
            v8h lo = *(const v8h*)&sA[arow * LDT + lhalf * 8];
            v8h hi = *(const v8h*)&sA[arow * LDT + lhalf * 8 + 16];
            af[t] = __builtin_shufflevector(lo, hi,
                0,1,2,3,4,5,6,7,8,9,10,11,12,13,14,15);
            // B 32x16 layout: lane = column N, K = {8*lhalf..+7, 16+8*lhalf..+7}
            int nrow = wn * 32 + t * 16 + l16;
            v8h blo = *(const v8h*)&sB[nrow * LDT + lhalf * 8];
            v8h bhi = *(const v8h*)&sB[nrow * LDT + lhalf * 8 + 16];
            bf[t] = __builtin_shufflevector(blo, bhi,
                0,1,2,3,4,5,6,7,8,9,10,11,12,13,14,15);
        }
#pragma unroll
        for (int tm = 0; tm < 2; ++tm)
#pragma unroll
            for (int tn = 0; tn < 2; ++tn)
                acc[tm][tn] = __builtin_amdgcn_wmma_f32_16x16x32_f16(
                    false, af[tm], false, bf[tn],
                    (short)0, acc[tm][tn], false, false);
    };

#if HAVE_ASYNC_LDS
    auto issue = [&](int buf, int k0) {
        async_copy_b128(gA0 + k0, &lA[buf][ar0 * LDT + ac]);
        async_copy_b128(gA1 + k0, &lA[buf][ar1 * LDT + ac]);
        async_copy_b128(gB  + k0, &lB[buf][ar0 * LDT + ac]);
    };
    issue(0, 0);
    issue(1, BK);
    int p = 0;
    for (int k0 = 0; k0 < KD; k0 += BK) {
        if (k0 + 2 * BK < KD) {        // prefetch 2 K-steps ahead
            issue((p + 2) & 3, k0 + 2 * BK);
            wait_asynccnt<6>();        // 3 ops/tile: leave 2 tiles in flight
        } else if (k0 + BK < KD) {
            wait_asynccnt<3>();
        } else {
            wait_asynccnt<0>();
        }
        __syncthreads();               // stage p fully landed for all waves
        compute_tile(lA[p], lB[p]);
        p = (p + 1) & 3;
    }
#else
    int p = 0;
    for (int k0 = 0; k0 < KD; k0 += BK) {
        *(uint4*)&lA[p][ar0 * LDT + ac] = *(const uint4*)(gA0 + k0);
        *(uint4*)&lA[p][ar1 * LDT + ac] = *(const uint4*)(gA1 + k0);
        *(uint4*)&lB[p][ar0 * LDT + ac] = *(const uint4*)(gB  + k0);
        __syncthreads();
        compute_tile(lA[p], lB[p]);
        p ^= 1;
    }
#endif

    // epilogue: C/D layout — VGPR r: M = r + 8*(lane/16), N = lane%16
#pragma unroll
    for (int tm = 0; tm < 2; ++tm) {
#pragma unroll
        for (int tn = 0; tn < 2; ++tn) {
            int col = n0 + wn * 32 + tn * 16 + l16;
            float bv = bias[col];
#pragma unroll
            for (int r = 0; r < 8; ++r) {
                int row = m0 + wm * 32 + tm * 16 + r + 8 * lhalf;
                float v = acc[tm][tn][r] + bv;
                if (RELU) v = v > 0.f ? v : 0.f;
                C[(size_t)row * ND + col] = (OutT)v;
            }
        }
    }
}

// ---------------------------------------------------------------------------
// z = mu + eps * exp(0.5*logvar); write z and z/||z|| (row L2 norm, L=128)
// ---------------------------------------------------------------------------
__global__ __launch_bounds__(128) void latent_kernel(
    const float* __restrict__ o4, const float* __restrict__ eps,
    float* __restrict__ zout, float* __restrict__ znorm)
{
    int n = blockIdx.x;
    int l = threadIdx.x;
    float mu = o4[(size_t)n * 256 + l];
    float lv = o4[(size_t)n * 256 + 128 + l];
    float z  = fmaf(eps[(size_t)n * 128 + l], expf(0.5f * lv), mu);

    float ss = z * z;
#pragma unroll
    for (int off = 16; off > 0; off >>= 1) ss += __shfl_xor(ss, off, 32);
    __shared__ float p[4];
    int w = l >> 5, ln = l & 31;
    if (ln == 0) p[w] = ss;
    __syncthreads();
    float inv = rsqrtf(p[0] + p[1] + p[2] + p[3]);
    zout [(size_t)n * 128 + l] = z;
    znorm[(size_t)n * 128 + l] = z * inv;
}

// ---------------------------------------------------------------------------
// gathered contrastive loss; one wave / sample; deterministic block partials
// ---------------------------------------------------------------------------
__global__ __launch_bounds__(256) void loss_kernel(
    const float* __restrict__ zm, const float* __restrict__ zd,
    const int* __restrict__ ts, float* __restrict__ parts, int S)
{
    int wave = threadIdx.x >> 5;
    int lane = threadIdx.x & 31;
    int s = blockIdx.x * 8 + wave;
    float contrib = 0.f;
    if (s < S) {
        int i  = ts[s * 4 + 0];
        int j  = ts[s * 4 + 1];
        int jh = ts[s * 4 + 3];
        const float4 a  = *(const float4*)&zm[(size_t)i  * 128 + lane * 4];
        const float4 bj = *(const float4*)&zd[(size_t)j  * 128 + lane * 4];
        const float4 bh = *(const float4*)&zd[(size_t)jh * 128 + lane * 4];
        float dij = a.x*bj.x + a.y*bj.y + a.z*bj.z + a.w*bj.w;
        float dih = a.x*bh.x + a.y*bh.y + a.z*bh.z + a.w*bh.w;
#pragma unroll
        for (int off = 16; off > 0; off >>= 1) {
            dij += __shfl_xor(dij, off, 32);
            dih += __shfl_xor(dih, off, 32);
        }
        if (lane == 0) {
            float loss1 = log1pf(expf(dih - dij));            // TAU = 1
            float loss2 = 0.5f * fabsf(dij - 1.f) + dih + 1.f;
            contrib = loss1 + loss2;
        }
    }
    __shared__ float p[8];
    if (lane == 0) p[wave] = contrib;
    __syncthreads();
    if (threadIdx.x == 0) {
        float t = 0.f;
        for (int w = 0; w < 8; ++w) t += p[w];
        parts[blockIdx.x] = t;
    }
}

__global__ __launch_bounds__(256) void reduce_loss(
    const float* __restrict__ parts, int n, float invS, float* __restrict__ out)
{
    float s = 0.f;
    for (int i = threadIdx.x; i < n; i += 256) s += parts[i];
#pragma unroll
    for (int off = 16; off > 0; off >>= 1) s += __shfl_xor(s, off, 32);
    __shared__ float p[8];
    if ((threadIdx.x & 31) == 0) p[threadIdx.x >> 5] = s;
    __syncthreads();
    if (threadIdx.x == 0) {
        float t = 0.f;
        for (int w = 0; w < 8; ++w) t += p[w];
        *out = t * invS;
    }
}

// ---------------------------------------------------------------------------
extern "C" void kernel_launch(void* const* d_in, const int* in_sizes, int n_in,
                              void* d_out, int out_size, void* d_ws, size_t ws_size,
                              hipStream_t stream)
{
    (void)in_sizes; (void)n_in; (void)out_size; (void)ws_size;
    const int Nr = 4096, D = 2048, L = 128, S = 32768;

    const float* feat[2] = {(const float*)d_in[0], (const float*)d_in[1]};
    const float* eps [2] = {(const float*)d_in[2], (const float*)d_in[3]};
    const float* W[2][4] = {
        {(const float*)d_in[4],  (const float*)d_in[6],  (const float*)d_in[8],  (const float*)d_in[10]},
        {(const float*)d_in[12], (const float*)d_in[14], (const float*)d_in[16], (const float*)d_in[18]}};
    const float* B[2][4] = {
        {(const float*)d_in[5],  (const float*)d_in[7],  (const float*)d_in[9],  (const float*)d_in[11]},
        {(const float*)d_in[13], (const float*)d_in[15], (const float*)d_in[17], (const float*)d_in[19]}};
    const int* ts = (const int*)d_in[20];

    float* out  = (float*)d_out;
    float* loss = out;
    float* zlat[2] = {out + 1, out + 1 + (size_t)Nr * L};

    const int ldim[5] = {D, 512, 256, 64, 2 * L};

    char* ws = (char*)d_ws;
    size_t off = 0;
    auto alloc = [&](size_t bytes) -> void* {
        void* p = ws + off;
        off += (bytes + 255) & ~(size_t)255;
        return p;
    };

    _Float16* xh[2];
    _Float16* Wh[2][4];
    _Float16* h [2][3];
    float*    o4[2];
    float*    zn[2];
    for (int e = 0; e < 2; ++e) {
        xh[e] = (_Float16*)alloc((size_t)Nr * D * 2);
        for (int l = 0; l < 4; ++l)
            Wh[e][l] = (_Float16*)alloc((size_t)ldim[l + 1] * ldim[l] * 2);
        for (int l = 0; l < 3; ++l)
            h[e][l] = (_Float16*)alloc((size_t)Nr * ldim[l + 1] * 2);
        o4[e] = (float*)alloc((size_t)Nr * 2 * L * 4);
        zn[e] = (float*)alloc((size_t)Nr * L * 4);
    }
    float* parts = (float*)alloc((size_t)(S / 8) * 4);

    for (int e = 0; e < 2; ++e) {
        int n8 = Nr * D / 8;
        f32_to_f16_kernel<<<(n8 + 255) / 256, 256, 0, stream>>>(feat[e], xh[e], n8);
        for (int l = 0; l < 4; ++l) {
            int m8 = ldim[l + 1] * ldim[l] / 8;
            f32_to_f16_kernel<<<(m8 + 255) / 256, 256, 0, stream>>>(W[e][l], Wh[e][l], m8);
        }
    }

    dim3 blk(256);
    for (int e = 0; e < 2; ++e) {
        gemm_bias_act<2048, 512, true,  _Float16><<<dim3(512 / BN, Nr / BM), blk, 0, stream>>>(
            xh[e],   Wh[e][0], B[e][0], h[e][0], Nr);
        gemm_bias_act< 512, 256, true,  _Float16><<<dim3(256 / BN, Nr / BM), blk, 0, stream>>>(
            h[e][0], Wh[e][1], B[e][1], h[e][1], Nr);
        gemm_bias_act< 256,  64, true,  _Float16><<<dim3( 64 / BN, Nr / BM), blk, 0, stream>>>(
            h[e][1], Wh[e][2], B[e][2], h[e][2], Nr);
        gemm_bias_act<  64, 256, false, float   ><<<dim3(256 / BN, Nr / BM), blk, 0, stream>>>(
            h[e][2], Wh[e][3], B[e][3], o4[e],   Nr);
        latent_kernel<<<Nr, 128, 0, stream>>>(o4[e], eps[e], zlat[e], zn[e]);
    }

    loss_kernel<<<S / 8, 256, 0, stream>>>(zn[0], zn[1], ts, parts, S);
    reduce_loss<<<1, 256, 0, stream>>>(parts, S / 8, 1.0f / (float)S, loss);
}
